// AttnProcessor_87660282511526
// MI455X (gfx1250) — matrix operand order repaired and checked by reading, not compiled, not run on hardware
//
#include <hip/hip_runtime.h>
#include <math.h>

// ---------------------------------------------------------------------------
// CDNA5 (gfx1250, wave32) fused attention processor
//   X[B,S,D] @ {Wq,Wk,Wv} -> flash attention -> @ Wo + bo
// Matmuls via v_wmma_f32_16x16x32_f16; all global->LDS staging uses the
// gfx1250 async-to-LDS path (ASYNCcnt) with double buffering.
// ---------------------------------------------------------------------------

typedef _Float16 v16h __attribute__((ext_vector_type(16)));
typedef _Float16 v8h  __attribute__((ext_vector_type(8)));
typedef float    v8f  __attribute__((ext_vector_type(8)));

#define S_LEN   2048
#define DMODEL  1280
#define NHEADS  20
#define HDIM    64
#define MROWS   4096   // B * S

__device__ __forceinline__ v16h join16(v8h lo, v8h hi) {
    union { v16h v; v8h h[2]; } u;
    u.h[0] = lo; u.h[1] = hi;
    return u.v;
}

// 16B async copy: global -> LDS, tracked by ASYNCcnt.
__device__ __forceinline__ void async_b128(unsigned lds_off, const void* gptr) {
    asm volatile("global_load_async_to_lds_b128 %0, %1, off"
                 :: "v"(lds_off), "v"((unsigned long long)(uintptr_t)gptr)
                 : "memory");
}
// generic (flat) pointer to LDS -> 32-bit LDS offset (low 32 bits of flat addr)
#define LDS_OFF(p) ((unsigned)(uintptr_t)(p))
#define WAIT_ASYNC(n) asm volatile("s_wait_asynccnt " #n ::: "memory")

// --------------------------- conversion kernels ----------------------------

__global__ __launch_bounds__(256)
void cvt_f32_to_f16(const float* __restrict__ in, _Float16* __restrict__ out, int n4) {
    int i = blockIdx.x * blockDim.x + threadIdx.x;
    if (i < n4) {
        float4 v = ((const float4*)in)[i];
        union { _Float16 h[4]; uint2 u; } pk;
        pk.h[0] = (_Float16)v.x; pk.h[1] = (_Float16)v.y;
        pk.h[2] = (_Float16)v.z; pk.h[3] = (_Float16)v.w;
        ((uint2*)out)[i] = pk.u;
    }
}

// W[k][n] f32  ->  WT[n][k] f16   (both DMODEL x DMODEL)
__global__ __launch_bounds__(256)
void cvt_transpose_w(const float* __restrict__ W0, const float* __restrict__ W1,
                     const float* __restrict__ W2, const float* __restrict__ W3,
                     _Float16* __restrict__ O0, _Float16* __restrict__ O1,
                     _Float16* __restrict__ O2, _Float16* __restrict__ O3) {
    const float* W = (blockIdx.z == 0) ? W0 : (blockIdx.z == 1) ? W1
                   : (blockIdx.z == 2) ? W2 : W3;
    _Float16* O    = (blockIdx.z == 0) ? O0 : (blockIdx.z == 1) ? O1
                   : (blockIdx.z == 2) ? O2 : O3;
    __shared__ _Float16 t[32][33];
    const int tx = threadIdx.x, ty = threadIdx.y;
    const int n0 = blockIdx.x * 32, k0 = blockIdx.y * 32;
#pragma unroll
    for (int yy = 0; yy < 32; yy += 8)
        t[ty + yy][tx] = (_Float16)W[(size_t)(k0 + ty + yy) * DMODEL + (n0 + tx)];
    __syncthreads();
#pragma unroll
    for (int yy = 0; yy < 32; yy += 8)
        O[(size_t)(n0 + ty + yy) * DMODEL + (k0 + tx)] = t[tx][ty + yy];
}

// ----------------------- shared GEMM mainloop (128x128) --------------------
// A[4096 x 1280] f16 row-major, B = W^T f16 [N][K] row-major.
// 256 threads = 8 waves; wave computes 64x32 (4x2 wmma tiles).
// Double-buffered async staging: copy of tile k+1 overlaps wmma of tile k.

__device__ __forceinline__ void gemm_issue_stage(const _Float16* __restrict__ A,
                                                 const _Float16* __restrict__ B,
                                                 _Float16* As, _Float16* Bs,
                                                 int mBase, int nBase, int kb, int tid) {
    const int srow = tid >> 1;           // 0..127
    const int scol = (tid & 1) * 16;     // 0 / 16 halves
    const _Float16* ga = A + (size_t)(mBase + srow) * DMODEL + kb + scol;
    const _Float16* gb = B + (size_t)(nBase + srow) * DMODEL + kb + scol;
    const unsigned la = LDS_OFF(&As[srow * 40 + scol]);
    const unsigned lb = LDS_OFF(&Bs[srow * 40 + scol]);
    async_b128(la,      ga);
    async_b128(la + 16, ga + 8);
    async_b128(lb,      gb);
    async_b128(lb + 16, gb + 8);
}

__device__ __forceinline__ void gemm_core(const _Float16* __restrict__ A,
                                          const _Float16* __restrict__ B,
                                          _Float16 (*As)[128 * 40],
                                          _Float16 (*Bs)[128 * 40],
                                          int mBase, int nBase, int tid,
                                          v8f acc[4][2]) {
    const int lane = tid & 31, wid = tid >> 5;
    const int hi   = (lane >> 4) & 1;
    const int l15  = lane & 15;
    const int waveM = (wid >> 2) * 64;
    const int waveN = (wid & 3) * 32;
    const int kb0 = hi ? 8 : 0;
    const int kbB = hi ? 16 : 0;

    gemm_issue_stage(A, B, As[0], Bs[0], mBase, nBase, 0, tid);
    int cur = 0;
    for (int kb = 0; kb < DMODEL; kb += 32) {
        if (kb + 32 < DMODEL) {
            gemm_issue_stage(A, B, As[cur ^ 1], Bs[cur ^ 1], mBase, nBase, kb + 32, tid);
            WAIT_ASYNC(4);   // retire previous stage; next stage stays in flight
        } else {
            WAIT_ASYNC(0);
        }
        __syncthreads();

        v16h afr[4], bfr[2];
#pragma unroll
        for (int i = 0; i < 4; i++) {
            const _Float16* p = &As[cur][(waveM + i * 16 + l15) * 40 + kb0];
            afr[i] = join16(*(const v8h*)p, *(const v8h*)(p + 16));
        }
#pragma unroll
        for (int j = 0; j < 2; j++) {
            const _Float16* p = &Bs[cur][(waveN + j * 16 + l15) * 40 + kb0];
            bfr[j] = join16(*(const v8h*)p, *(const v8h*)(p + 16));
        }
#pragma unroll
        for (int i = 0; i < 4; i++)
#pragma unroll
            for (int j = 0; j < 2; j++)
                acc[i][j] = __builtin_amdgcn_wmma_f32_16x16x32_f16(
                    false, afr[i], false, bfr[j], (short)0, acc[i][j], false, false);
        __syncthreads();
        cur ^= 1;
    }
}

// --------------------------- QKV projection GEMM ---------------------------
// blockIdx.z selects Wq/Wk/Wv.  Q,K stored [B,H,S,hd]; V stored [B,H,hd,S].

__global__ __launch_bounds__(256)
void gemm_qkv(const _Float16* __restrict__ Xh,
              const _Float16* __restrict__ WTq,
              const _Float16* __restrict__ WTk,
              const _Float16* __restrict__ WTv,
              _Float16* __restrict__ Qh,
              _Float16* __restrict__ Kh,
              _Float16* __restrict__ Vt) {
    const int which = blockIdx.z;
    const _Float16* WT = (which == 0) ? WTq : (which == 1) ? WTk : WTv;

    __shared__ __align__(16) _Float16 As[2][128 * 40];
    __shared__ __align__(16) _Float16 Bs[2][128 * 40];

    const int tid  = threadIdx.x;
    const int lane = tid & 31, wid = tid >> 5;
    const int hi   = (lane >> 4) & 1;
    const int l15  = lane & 15;
    const int mBase = blockIdx.y * 128;
    const int nBase = blockIdx.x * 128;
    const int waveM = (wid >> 2) * 64;
    const int waveN = (wid & 3) * 32;

    v8f acc[4][2] = {};
    gemm_core(Xh, WT, As, Bs, mBase, nBase, tid, acc);

    // epilogue: scatter into head-major layouts (f16)
#pragma unroll
    for (int i = 0; i < 4; i++) {
        const int mg = mBase + waveM + i * 16 + hi * 8;   // + r
        const int b  = mg >> 11;
        const int s0 = mg & (S_LEN - 1);
#pragma unroll
        for (int j = 0; j < 2; j++) {
            const int ng = nBase + waveN + j * 16 + l15;
            const int h = ng >> 6, d = ng & 63;
            if (which == 2) {
                // V transposed [B,H,hd,S]: 8 consecutive s -> one b128 store
                union { _Float16 h8[8]; uint4 u; } pk;
#pragma unroll
                for (int r = 0; r < 8; r++) pk.h8[r] = (_Float16)acc[i][j][r];
                *(uint4*)&Vt[(((size_t)b * NHEADS + h) * HDIM + d) * S_LEN + s0] = pk.u;
            } else {
                _Float16* dst = (which == 0) ? Qh : Kh;
                const size_t base = (((size_t)b * NHEADS + h) * S_LEN + s0) * HDIM + d;
#pragma unroll
                for (int r = 0; r < 8; r++)
                    dst[base + (size_t)r * HDIM] = (_Float16)acc[i][j][r];
            }
        }
    }
}

// ------------------------------ flash attention ----------------------------
// grid(32 q-tiles, 40 b*h), 128 threads = 4 waves; wave owns 16 q rows.
// All 4 waves consume the same K/V block: stage cooperatively via async-to-LDS
// (double buffered).  Q pre-scaled by sm_scale*log2(e); softmax in exp2 domain.

__device__ __forceinline__ void attn_issue_stage(const _Float16* __restrict__ kB,
                                                 const _Float16* __restrict__ vB,
                                                 _Float16* Ks, _Float16* Vs,
                                                 int kt, int tid) {
    {   // K tile: 32 rows x 64 halves (rows = keys, cols = d)
        const int row = tid >> 2, chunk = (tid & 3) * 8;
        async_b128(LDS_OFF(&Ks[row * 72 + chunk]),
                   kB + (size_t)(kt + row) * HDIM + chunk);
    }
    {   // V tile: 64 rows x 32 halves (rows = d, cols = keys)
        const int row = tid >> 1, chunk = (tid & 1) * 8;
        async_b128(LDS_OFF(&Vs[row * 40 + chunk]),
                   vB + (size_t)row * S_LEN + kt + chunk);
    }
}

__global__ __launch_bounds__(128)
void attn_flash(const _Float16* __restrict__ Qh,
                const _Float16* __restrict__ Kh,
                const _Float16* __restrict__ Vt,
                _Float16* __restrict__ Oh) {
    __shared__ __align__(16) _Float16 Ks[2][32 * 72];
    __shared__ __align__(16) _Float16 Vs[2][64 * 40];
    __shared__ __align__(16) _Float16 Plds[4][16 * 40];

    const int tid  = threadIdx.x;
    const int lane = tid & 31, w = tid >> 5;
    const int hi   = (lane >> 4) & 1;
    const int l15  = lane & 15;
    const int bh = blockIdx.y;
    const int b = bh / NHEADS, h = bh % NHEADS;
    const int q0 = blockIdx.x * 64 + w * 16;

    const _Float16* qB = Qh + (size_t)bh * S_LEN * HDIM;
    const _Float16* kB = Kh + (size_t)bh * S_LEN * HDIM;
    const _Float16* vB = Vt + (size_t)bh * HDIM * S_LEN;

    const int kb0 = hi ? 8 : 0;
    const int kbB = hi ? 16 : 0;

    // preload Q fragments, folded scale = 1/sqrt(64) * log2(e)
    const _Float16 qs = (_Float16)(0.125f * 1.44269504088896340736f);
    v16h aq[2];
#pragma unroll
    for (int ks = 0; ks < 2; ks++) {
        const _Float16* p = qB + (size_t)(q0 + l15) * HDIM + ks * 32 + kb0;
        v8h lo = *(const v8h*)p;
        v8h hh = *(const v8h*)(p + 16);
        aq[ks] = join16(lo * qs, hh * qs);
    }

    float mrow[8], lrow[8];
#pragma unroll
    for (int r = 0; r < 8; r++) { mrow[r] = -__builtin_inff(); lrow[r] = 0.0f; }
    v8f accO[4] = {};

    attn_issue_stage(kB, vB, Ks[0], Vs[0], 0, tid);
    int cur = 0;
    for (int kt = 0; kt < S_LEN; kt += 32) {
        if (kt + 32 < S_LEN) {
            attn_issue_stage(kB, vB, Ks[cur ^ 1], Vs[cur ^ 1], kt + 32, tid);
            WAIT_ASYNC(2);
        } else {
            WAIT_ASYNC(0);
        }
        __syncthreads();

        // ---- scores: S = (Q*scale) K^T  (16 x 32, exp2 domain) ----
        v8f sc[2] = {};
#pragma unroll
        for (int j = 0; j < 2; j++) {
#pragma unroll
            for (int ks = 0; ks < 2; ks++) {
                const _Float16* p = &Ks[cur][(j * 16 + l15) * 72 + ks * 32 + kb0];
                v16h bk = join16(*(const v8h*)p, *(const v8h*)(p + 16));
                sc[j] = __builtin_amdgcn_wmma_f32_16x16x32_f16(
                    false, aq[ks], false, bk, (short)0, sc[j], false, false);
            }
        }
        // ---- online softmax (rows live across the 16-lane half-wave) ----
#pragma unroll
        for (int r = 0; r < 8; r++) {
            float bm = fmaxf(sc[0][r], sc[1][r]);
#pragma unroll
            for (int off = 1; off < 16; off <<= 1)
                bm = fmaxf(bm, __shfl_xor(bm, off, 32));
            const float mn   = fmaxf(mrow[r], bm);
            const float corr = exp2f(mrow[r] - mn);
            const float p0 = exp2f(sc[0][r] - mn);
            const float p1 = exp2f(sc[1][r] - mn);
            sc[0][r] = p0; sc[1][r] = p1;
            float ps = p0 + p1;
#pragma unroll
            for (int off = 1; off < 16; off <<= 1)
                ps += __shfl_xor(ps, off, 32);
            lrow[r] = lrow[r] * corr + ps;
            mrow[r] = mn;
#pragma unroll
            for (int n = 0; n < 4; n++) accO[n][r] *= corr;
        }
        // ---- redistribute P: C-layout -> A-layout via per-wave LDS ----
        {
            _Float16* pw = &Plds[w][(hi * 8) * 40 + l15];
#pragma unroll
            for (int r = 0; r < 8; r++) {
                pw[r * 40]      = (_Float16)sc[0][r];
                pw[r * 40 + 16] = (_Float16)sc[1][r];
            }
        }
        const _Float16* pr = &Plds[w][l15 * 40 + kb0];
        v16h pfrag = join16(*(const v8h*)pr, *(const v8h*)(pr + 16));
        // ---- O += P V  (V tile transposed: contiguous K per d-row) ----
#pragma unroll
        for (int n = 0; n < 4; n++) {
            const _Float16* p = &Vs[cur][(n * 16 + l15) * 40 + kb0];
            v16h bv = join16(*(const v8h*)p, *(const v8h*)(p + 16));
            accO[n] = __builtin_amdgcn_wmma_f32_16x16x32_f16(
                false, pfrag, false, bv, (short)0, accO[n], false, false);
        }
        __syncthreads();
        cur ^= 1;
    }

    // ---- normalize and store O as [B,S,H*hd] f16 ----
#pragma unroll
    for (int r = 0; r < 8; r++) {
        const float inv = 1.0f / lrow[r];
        const int s = q0 + hi * 8 + r;
        const size_t base = ((size_t)b * S_LEN + s) * DMODEL + h * HDIM;
#pragma unroll
        for (int n = 0; n < 4; n++)
            Oh[base + n * 16 + l15] = (_Float16)(accO[n][r] * inv);
    }
}

// ------------------------------ output GEMM --------------------------------
// out[4096 x 1280] f32 = Oh[4096 x 1280] * Wo + bo ; WoT f16 [N][K].

__global__ __launch_bounds__(256)
void gemm_out(const _Float16* __restrict__ Ah,
              const _Float16* __restrict__ WT,
              const float* __restrict__ bias,
              float* __restrict__ out) {
    __shared__ __align__(16) _Float16 As[2][128 * 40];
    __shared__ __align__(16) _Float16 Bs[2][128 * 40];

    const int tid  = threadIdx.x;
    const int lane = tid & 31, wid = tid >> 5;
    const int hi   = (lane >> 4) & 1;
    const int l15  = lane & 15;
    const int mBase = blockIdx.y * 128;
    const int nBase = blockIdx.x * 128;
    const int waveM = (wid >> 2) * 64;
    const int waveN = (wid & 3) * 32;

    v8f acc[4][2] = {};
    gemm_core(Ah, WT, As, Bs, mBase, nBase, tid, acc);

#pragma unroll
    for (int j = 0; j < 2; j++) {
        const int ng = nBase + waveN + j * 16 + l15;
        const float bv = bias[ng];
#pragma unroll
        for (int i = 0; i < 4; i++) {
            const int mg = mBase + waveM + i * 16 + hi * 8;
#pragma unroll
            for (int r = 0; r < 8; r++)
                out[(size_t)(mg + r) * DMODEL + ng] = acc[i][j][r] + bv;
        }
    }
}

// ------------------------------ host launcher ------------------------------

extern "C" void kernel_launch(void* const* d_in, const int* in_sizes, int n_in,
                              void* d_out, int out_size, void* d_ws, size_t ws_size,
                              hipStream_t stream) {
    (void)in_sizes; (void)n_in; (void)out_size; (void)ws_size;

    const float* X  = (const float*)d_in[0];
    const float* Wq = (const float*)d_in[1];
    const float* Wk = (const float*)d_in[2];
    const float* Wv = (const float*)d_in[3];
    const float* Wo = (const float*)d_in[4];
    const float* bo = (const float*)d_in[5];
    float* out = (float*)d_out;

    // workspace carve-up (~63 MB total)
    char* ws = (char*)d_ws;
    size_t off = 0;
    auto take = [&](size_t bytes) {
        void* p = ws + off;
        off += (bytes + 255) & ~(size_t)255;
        return p;
    };
    _Float16* Xh  = (_Float16*)take((size_t)MROWS * DMODEL * 2);
    _Float16* WqT = (_Float16*)take((size_t)DMODEL * DMODEL * 2);
    _Float16* WkT = (_Float16*)take((size_t)DMODEL * DMODEL * 2);
    _Float16* WvT = (_Float16*)take((size_t)DMODEL * DMODEL * 2);
    _Float16* WoT = (_Float16*)take((size_t)DMODEL * DMODEL * 2);
    _Float16* Qh  = (_Float16*)take((size_t)MROWS * DMODEL * 2);
    _Float16* Kh  = (_Float16*)take((size_t)MROWS * DMODEL * 2);
    _Float16* Vt  = (_Float16*)take((size_t)MROWS * DMODEL * 2);
    _Float16* Oh  = (_Float16*)take((size_t)MROWS * DMODEL * 2);

    // 1) precision conversion (+ weight transposes)
    {
        int n4 = MROWS * DMODEL / 4;
        cvt_f32_to_f16<<<(n4 + 255) / 256, 256, 0, stream>>>(X, Xh, n4);
        dim3 blk(32, 8);
        cvt_transpose_w<<<dim3(40, 40, 4), blk, 0, stream>>>(
            Wq, Wk, Wv, Wo, WqT, WkT, WvT, WoT);
    }
    // 2) fused QKV projection (grid.z selects weight)
    gemm_qkv<<<dim3(DMODEL / 128, MROWS / 128, 3), 256, 0, stream>>>(
        Xh, WqT, WkT, WvT, Qh, Kh, Vt);
    // 3) flash attention
    attn_flash<<<dim3(S_LEN / 64, 2 * NHEADS), 128, 0, stream>>>(Qh, Kh, Vt, Oh);
    // 4) output projection + bias
    gemm_out<<<dim3(DMODEL / 128, MROWS / 128), 256, 0, stream>>>(Oh, WoT, bo, out);
}
